// MTLNetwork_13718125543740
// MI455X (gfx1250) — matrix operand order repaired and checked by reading, not compiled, not run on hardware
//
#include <hip/hip_runtime.h>

// ---------------------------------------------------------------------------
// MTL (PLE / multi-gate MoE) fused forward for gfx1250 (MI455X).
// f16 WMMA (V_WMMA_F32_16X16X32_F16) + fp32 accumulation, fully fused.
//  - A-fragments for x hoisted into VGPRs once per kernel.
//  - Weight staging via GLOBAL_LOAD_ASYNC_TO_LDS_B128 (ASYNCcnt) with a
//    3-slot LDS ring: next matrix is prefetched during current phase compute,
//    synced with s_wait_asynccnt + workgroup barrier.
//  - Wave-private LDS staging needs no extra barriers (same-wave DS in-order).
// ---------------------------------------------------------------------------

typedef __attribute__((ext_vector_type(16))) _Float16 v16h;
typedef __attribute__((ext_vector_type(8)))  _Float16 v8h;
typedef __attribute__((ext_vector_type(8)))  float    v8f;

#define BDIM   256
#define SW     136          // padded LDS row stride (halfs)
#define NB     65536
#define DIM    128
#define MAT    16384        // 128*128
#define RSQRT_D 0.17677669529663687f   // 1/sqrt(32)

#define Z8 {0.f,0.f,0.f,0.f,0.f,0.f,0.f,0.f}

#if defined(__has_builtin)
#  if __has_builtin(__builtin_amdgcn_global_load_async_to_lds_b128)
#    define HAVE_ASYNC_LDS 1
#  else
#    define HAVE_ASYNC_LDS 0
#  endif
#else
#  define HAVE_ASYNC_LDS 0
#endif

#if HAVE_ASYNC_LDS
typedef int v4i_vs __attribute__((vector_size(16)));
typedef __attribute__((address_space(1))) v4i_vs* gas_ptr;   // global int4*
typedef __attribute__((address_space(3))) v4i_vs* las_ptr;   // LDS int4*
#endif

struct PrepParams {
  const float* W[3][5];     // [group][Wq,Wk,Wv,Wo,Wfc]
  const float* gW1[2];
  const float* gW2[2];
  _Float16* wsW;            // 60 * 16384 halfs, transposed [out][in]
  _Float16* wsG1;           // 2 * 16384
  _Float16* wsG2;           // 2 * 2048 ([16][128], rows 8..15 zero)
};

struct KParams {
  const float* x;
  const float* bq[3]; const float* bk[3]; const float* bv[3];
  const float* bo[3]; const float* bfc[3];
  const float* gam[3]; const float* bet[3]; const float* mea[3]; const float* var[3];
  const float* gb1[2]; const float* gb2[2];
  const _Float16* wsW; const _Float16* wsG1; const _Float16* wsG2;
  float* out1; float* out2;
};

// ---- prep: f32 -> f16, transpose to [out][in] ------------------------------
__global__ __launch_bounds__(BDIM) void prep_kernel(PrepParams P) {
  const int b = blockIdx.x, tid = threadIdx.x;
  if (b < 60) {
    const int e = b / 5, m = b % 5;
    const int g = e >> 2, lc = e & 3;
    const float* src = P.W[g][m] + (size_t)lc * MAT;
    _Float16* dst = P.wsW + (size_t)b * MAT;
    for (int i = tid; i < MAT; i += BDIM) {
      const int out = i >> 7, in = i & 127;
      dst[i] = (_Float16)src[in * DIM + out];
    }
  } else {
    const int g = b - 60;
    const float* s1p = P.gW1[g];
    _Float16* d1 = P.wsG1 + (size_t)g * MAT;
    for (int i = tid; i < MAT; i += BDIM) {
      const int out = i >> 7, in = i & 127;
      d1[i] = (_Float16)s1p[in * DIM + out];
    }
    const float* s2p = P.gW2[g];
    _Float16* d2 = P.wsG2 + (size_t)g * 2048;
    for (int i = tid; i < 2048; i += BDIM) {
      const int out = i >> 7, in = i & 127;
      d2[i] = (out < 8) ? (_Float16)s2p[in * 8 + out] : (_Float16)0.0f;
    }
  }
}

// ---- fragment helpers (ISA 7.12.2 layouts) ---------------------------------
__device__ __forceinline__ v8f wmma_f16(v16h a, v16h b, v8f c) {
  return __builtin_amdgcn_wmma_f32_16x16x32_f16(false, a, false, b, (short)0, c,
                                                false, false);
}

__device__ __forceinline__ v16h loadA(const _Float16* tile, int kk, int lane) {
  const int m = lane & 15, hbk = (lane >> 4) & 1;
  const _Float16* p = tile + m * SW + kk * 32 + hbk * 8;
  v8h lo = *(const v8h*)p;
  v8h hi = *(const v8h*)(p + 16);
  return __builtin_shufflevector(lo, hi, 0,1,2,3,4,5,6,7,8,9,10,11,12,13,14,15);
}

__device__ __forceinline__ v16h loadB(const _Float16* wtile, int nt, int kk, int lane) {
  const int n = lane & 15, hbk = (lane >> 4) & 1;
  const _Float16* p = wtile + (nt * 16 + n) * SW + kk * 32 + hbk * 8;
  v8h lo = *(const v8h*)p;
  v8h hi = *(const v8h*)(p + 16);
  return __builtin_shufflevector(lo, hi, 0,1,2,3,4,5,6,7,8,9,10,11,12,13,14,15);
}

// cooperative stage: global f16 [rows][128] -> LDS [rows][SW]
__device__ __forceinline__ void stageRows(const _Float16* src, _Float16* dst,
                                          int rows, int tid) {
  const int chunks = rows << 4;
#if HAVE_ASYNC_LDS
  for (int c = tid; c < chunks; c += BDIM) {
    const int r = c >> 4, co = (c & 15) << 3;
    __builtin_amdgcn_global_load_async_to_lds_b128(
        (gas_ptr)(src + (r << 7) + co),
        (las_ptr)(dst + r * SW + co), 0, 0);
  }
#else
  for (int c = tid; c < chunks; c += BDIM) {
    const int r = c >> 4, co = (c & 15) << 3;
    *(v8h*)(dst + r * SW + co) = *(const v8h*)(src + (r << 7) + co);
  }
#endif
}

__device__ __forceinline__ void waitStage() {
#if HAVE_ASYNC_LDS
#  if __has_builtin(__builtin_amdgcn_s_wait_asynccnt)
  __builtin_amdgcn_s_wait_asynccnt(0);
#  else
  asm volatile("s_wait_asynccnt 0x0" ::: "memory");
#  endif
#endif
}

__device__ __forceinline__ float rsum16(float v) {
  v += __shfl_xor(v, 1); v += __shfl_xor(v, 2);
  v += __shfl_xor(v, 4); v += __shfl_xor(v, 8);
  return v;
}
__device__ __forceinline__ float rmax16(float v) {
  v = fmaxf(v, __shfl_xor(v, 1)); v = fmaxf(v, __shfl_xor(v, 2));
  v = fmaxf(v, __shfl_xor(v, 4)); v = fmaxf(v, __shfl_xor(v, 8));
  return v;
}

// ---- main fused kernel -----------------------------------------------------
__global__ __launch_bounds__(BDIM, 1) void mtl_kernel(KParams P) {
  const int tid  = threadIdx.x;
  const int lane = tid & 31;
  const int wv   = tid >> 5;
  const int l15  = lane & 15;
  const int hb   = lane >> 4;
  const int rowBase = blockIdx.x * 128 + wv * 16;

  __shared__ __align__(16) _Float16 xbuf[8][16 * SW];
  __shared__ __align__(16) _Float16 sA[8][16 * SW];     // v/o staging (wave-private)
  __shared__ __align__(16) _Float16 sB[8][16 * SW];     // oo staging (wave-private)
  __shared__ __align__(16) _Float16 wb[3][128 * SW];    // weight slot ring
  __shared__ __align__(16) _Float16 gbf[2][16 * SW];    // gate W2 (padded)

  _Float16* myx = xbuf[wv];
  _Float16* myA = sA[wv];
  _Float16* myB = sB[wv];

  // x rows -> f16 LDS (wave-private), then hoist the 4 A-fragments forever.
  #pragma unroll
  for (int r = 0; r < 16; ++r) {
    const float4 xv = *(const float4*)(P.x + (size_t)(rowBase + r) * DIM + lane * 4);
    _Float16* d = myx + r * SW + lane * 4;
    d[0] = (_Float16)xv.x; d[1] = (_Float16)xv.y;
    d[2] = (_Float16)xv.z; d[3] = (_Float16)xv.w;
  }
  v16h aX[4];
  #pragma unroll
  for (int kk = 0; kk < 4; ++kk) aX[kk] = loadA(myx, kk, lane);

  float o1[64], o2[64];
  #pragma unroll
  for (int i = 0; i < 64; ++i) { o1[i] = 0.f; o2[i] = 0.f; }

  // stream m0,m1 (gate W1s) + gate W2s
  stageRows(P.wsG1,        wb[0], 128, tid);
  stageRows(P.wsG1 + MAT,  wb[1], 128, tid);
  stageRows(P.wsG2,        gbf[0], 16, tid);
  stageRows(P.wsG2 + 2048, gbf[1], 16, tid);
  waitStage(); __syncthreads();

  // ---------------- gates: g = softmax(relu(x W1 + b1) W2 + b2) -------------
  float gate[2][8];
  #pragma unroll
  for (int g = 0; g < 2; ++g) {
    // prefetch stream m2 (Wq e0 -> slot 2) / m3 (Wk e0 -> slot 0)
    if (g == 0) stageRows(P.wsW,       wb[2], 128, tid);
    else        stageRows(P.wsW + MAT, wb[0], 128, tid);

    #pragma unroll
    for (int nt = 0; nt < 8; ++nt) {
      v8f acc = Z8;
      #pragma unroll
      for (int kk = 0; kk < 4; ++kk)
        acc = wmma_f16(aX[kk], loadB(wb[g], nt, kk, lane), acc);
      const float b1v = P.gb1[g][nt * 16 + l15];
      #pragma unroll
      for (int j = 0; j < 8; ++j)
        myA[(j + hb * 8) * SW + nt * 16 + l15] = (_Float16)fmaxf(acc[j] + b1v, 0.f);
    }
    v16h aH[4];
    #pragma unroll
    for (int kk = 0; kk < 4; ++kk) aH[kk] = loadA(myA, kk, lane);
    v8f lacc = Z8;
    #pragma unroll
    for (int kk = 0; kk < 4; ++kk)
      lacc = wmma_f16(aH[kk], loadB(gbf[g], 0, kk, lane), lacc);
    const bool valid = (l15 < 8);
    const float b2v = valid ? P.gb2[g][l15] : 0.f;
    #pragma unroll
    for (int j = 0; j < 8; ++j) {
      const float lg = lacc[j] + b2v;
      const float mx = rmax16(valid ? lg : -3.0e38f);
      const float ex = valid ? __expf(lg - mx) : 0.f;
      const float sm = rsum16(ex);
      gate[g][j] = ex / sm;
    }
    if (g == 0) __syncthreads();   // all waves done with slot 0 before Wk prefetch
  }
  waitStage(); __syncthreads();    // Wq,Wk of expert 0 staged

  // ---------------- expert loop --------------------------------------------
  #pragma unroll 1
  for (int e = 0; e < 12; ++e) {
    const int g = e >> 2, lc = e & 3;
    const _Float16* We = P.wsW + (size_t)e * 5 * MAT;
    const int sQ = (2 + 2 * e) % 3;        // slot of Wq (stream index 2+5e)
    const int sK = (sQ + 1) % 3;           // Wk
    const int sV = (sQ + 2) % 3;           // Wv
    const float* bqp = P.bq[g]  + lc * DIM;
    const float* bkp = P.bk[g]  + lc * DIM;
    const float* bvp = P.bv[g]  + lc * DIM;
    const float* bop = P.bo[g]  + lc * DIM;
    const float* bfp = P.bfc[g] + lc * DIM;
    const float* gmp = P.gam[g] + lc * DIM;
    const float* btp = P.bet[g] + lc * DIM;
    const float* mnp = P.mea[g] + lc * DIM;
    const float* vrp = P.var[g] + lc * DIM;

    // --- phase QK (slots sQ,sK); prefetch Wv -> sV ---
    stageRows(We + 2 * MAT, wb[sV], 128, tid);
    float sc[4][8];
    #pragma unroll
    for (int h = 0; h < 4; ++h)
      #pragma unroll
      for (int j = 0; j < 8; ++j) sc[h][j] = 0.f;

    #pragma unroll
    for (int nt = 0; nt < 8; ++nt) {
      v8f qa = Z8, ka = Z8;
      #pragma unroll
      for (int kk = 0; kk < 4; ++kk) {
        qa = wmma_f16(aX[kk], loadB(wb[sQ], nt, kk, lane), qa);
        ka = wmma_f16(aX[kk], loadB(wb[sK], nt, kk, lane), ka);
      }
      const float bqv = bqp[nt * 16 + l15];
      const float bkv = bkp[nt * 16 + l15];
      const int h = nt >> 1;
      #pragma unroll
      for (int j = 0; j < 8; ++j)
        sc[h][j] += rsum16((qa[j] + bqv) * (ka[j] + bkv));
    }
    float attn[4][8];
    #pragma unroll
    for (int j = 0; j < 8; ++j) {
      const float a0 = sc[0][j] * RSQRT_D, a1 = sc[1][j] * RSQRT_D;
      const float a2 = sc[2][j] * RSQRT_D, a3 = sc[3][j] * RSQRT_D;
      const float mx = fmaxf(fmaxf(a0, a1), fmaxf(a2, a3));
      const float e0 = __expf(a0 - mx), e1 = __expf(a1 - mx);
      const float e2 = __expf(a2 - mx), e3 = __expf(a3 - mx);
      const float inv = 1.f / (e0 + e1 + e2 + e3);
      attn[0][j] = e0 * inv; attn[1][j] = e1 * inv;
      attn[2][j] = e2 * inv; attn[3][j] = e3 * inv;
    }
    waitStage(); __syncthreads();

    // --- phase V (slot sV): o = attn*(x Wv + bv) -> myA; prefetch Wo -> sQ ---
    stageRows(We + 3 * MAT, wb[sQ], 128, tid);
    #pragma unroll
    for (int nt = 0; nt < 8; ++nt) {
      v8f va = Z8;
      #pragma unroll
      for (int kk = 0; kk < 4; ++kk)
        va = wmma_f16(aX[kk], loadB(wb[sV], nt, kk, lane), va);
      const float bvv = bvp[nt * 16 + l15];
      const int h = nt >> 1;
      #pragma unroll
      for (int j = 0; j < 8; ++j)
        myA[(j + hb * 8) * SW + nt * 16 + l15] =
            (_Float16)(attn[h][j] * (va[j] + bvv));
    }
    waitStage(); __syncthreads();

    // --- phase O (slot sQ holds Wo): oo = o Wo + bo -> myB; prefetch Wfc -> sK
    stageRows(We + 4 * MAT, wb[sK], 128, tid);
    v16h aO[4];
    #pragma unroll
    for (int kk = 0; kk < 4; ++kk) aO[kk] = loadA(myA, kk, lane);
    #pragma unroll
    for (int nt = 0; nt < 8; ++nt) {
      v8f oa = Z8;
      #pragma unroll
      for (int kk = 0; kk < 4; ++kk)
        oa = wmma_f16(aO[kk], loadB(wb[sQ], nt, kk, lane), oa);
      const float bov = bop[nt * 16 + l15];
      #pragma unroll
      for (int j = 0; j < 8; ++j)
        myB[(j + hb * 8) * SW + nt * 16 + l15] = (_Float16)(oa[j] + bov);
    }
    waitStage(); __syncthreads();

    // --- phase FC (slot sK holds Wfc); prefetch next Wq->sV, Wk->sQ ---
    if (e < 11) {
      stageRows(We + 5 * MAT, wb[sV], 128, tid);
      stageRows(We + 6 * MAT, wb[sQ], 128, tid);
    }
    float w1[8], w2[8];
    const int gi1 = (e < 8) ? e : -1;
    const int gi2 = (e < 4) ? e : ((e >= 8) ? e - 4 : -1);
    #pragma unroll
    for (int j = 0; j < 8; ++j) {
      const float a1 = __shfl(gate[0][j], (lane & 16) | (gi1 & 7));
      const float a2 = __shfl(gate[1][j], (lane & 16) | (gi2 & 7));
      w1[j] = (gi1 >= 0) ? a1 : 0.f;
      w2[j] = (gi2 >= 0) ? a2 : 0.f;
    }
    v16h aF[4];
    #pragma unroll
    for (int kk = 0; kk < 4; ++kk) aF[kk] = loadA(myB, kk, lane);
    #pragma unroll
    for (int nt = 0; nt < 8; ++nt) {
      v8f fa = Z8;
      #pragma unroll
      for (int kk = 0; kk < 4; ++kk)
        fa = wmma_f16(aF[kk], loadB(wb[sK], nt, kk, lane), fa);
      const int col = nt * 16 + l15;
      const float bfv   = bfp[col];
      const float scale = gmp[col] * rsqrtf(vrp[col] + 1e-5f);
      const float shift = btp[col] - mnp[col] * scale;
      #pragma unroll
      for (int j = 0; j < 8; ++j) {
        const float hh  = fmaxf(fa[j] + bfv, 0.f);
        const float hbn = fmaf(hh, scale, shift);
        o1[nt * 8 + j] = fmaf(w1[j], hbn, o1[nt * 8 + j]);
        o2[nt * 8 + j] = fmaf(w2[j], hbn, o2[nt * 8 + j]);
      }
    }
    waitStage(); __syncthreads();
  }

  // ---------------- store both task outputs --------------------------------
  #pragma unroll
  for (int nt = 0; nt < 8; ++nt)
    #pragma unroll
    for (int j = 0; j < 8; ++j) {
      const size_t idx = (size_t)(rowBase + j + hb * 8) * DIM + nt * 16 + l15;
      P.out1[idx] = o1[nt * 8 + j];
      P.out2[idx] = o2[nt * 8 + j];
    }
}

// ---------------------------------------------------------------------------
// Input index map (setup_inputs insertion order):
// 0: x
// 1..14:  shared:  Wq bq Wk bk Wv bv Wo bo Wfc bfc gamma beta mean var
// 15..28: task1, 29..42: task2
// 43..46: gate_shared (unused), 47..50: gate_task1, 51..54: gate_task2
// gate params order: W1 b1 W2 b2
// ---------------------------------------------------------------------------
extern "C" void kernel_launch(void* const* d_in, const int* in_sizes, int n_in,
                              void* d_out, int out_size, void* d_ws, size_t ws_size,
                              hipStream_t stream) {
  (void)in_sizes; (void)n_in; (void)out_size; (void)ws_size;

  _Float16* wsW  = (_Float16*)d_ws;
  _Float16* wsG1 = wsW + (size_t)60 * MAT;
  _Float16* wsG2 = wsG1 + (size_t)2 * MAT;

  PrepParams pp;
  for (int g = 0; g < 3; ++g) {
    const int base = 1 + g * 14;
    pp.W[g][0] = (const float*)d_in[base + 0];   // Wq
    pp.W[g][1] = (const float*)d_in[base + 2];   // Wk
    pp.W[g][2] = (const float*)d_in[base + 4];   // Wv
    pp.W[g][3] = (const float*)d_in[base + 6];   // Wo
    pp.W[g][4] = (const float*)d_in[base + 8];   // Wfc
  }
  pp.gW1[0] = (const float*)d_in[47]; pp.gW2[0] = (const float*)d_in[49];
  pp.gW1[1] = (const float*)d_in[51]; pp.gW2[1] = (const float*)d_in[53];
  pp.wsW = wsW; pp.wsG1 = wsG1; pp.wsG2 = wsG2;

  prep_kernel<<<62, BDIM, 0, stream>>>(pp);

  KParams kp;
  kp.x = (const float*)d_in[0];
  for (int g = 0; g < 3; ++g) {
    const int base = 1 + g * 14;
    kp.bq[g]  = (const float*)d_in[base + 1];
    kp.bk[g]  = (const float*)d_in[base + 3];
    kp.bv[g]  = (const float*)d_in[base + 5];
    kp.bo[g]  = (const float*)d_in[base + 7];
    kp.bfc[g] = (const float*)d_in[base + 9];
    kp.gam[g] = (const float*)d_in[base + 10];
    kp.bet[g] = (const float*)d_in[base + 11];
    kp.mea[g] = (const float*)d_in[base + 12];
    kp.var[g] = (const float*)d_in[base + 13];
  }
  kp.gb1[0] = (const float*)d_in[48]; kp.gb2[0] = (const float*)d_in[50];
  kp.gb1[1] = (const float*)d_in[52]; kp.gb2[1] = (const float*)d_in[54];
  kp.wsW = wsW; kp.wsG1 = wsG1; kp.wsG2 = wsG2;
  kp.out1 = (float*)d_out;
  kp.out2 = (float*)d_out + (size_t)NB * DIM;

  mtl_kernel<<<NB / 128, BDIM, 0, stream>>>(kp);
}
